// RCNNWLModel_24704651886894
// MI455X (gfx1250) — matrix operand order repaired and checked by reading, not compiled, not run on hardware
//
#include <hip/hip_runtime.h>
#include <hip/hip_bf16.h>

// ---------------- problem constants (from reference setup_inputs) ------------
#define B_   128
#define N_   200
#define M_   500
#define K_   10
#define AF_  82
#define BF_  6
#define H_   256
#define MN_  (B_ * N_)          // 25600 rows for all atom-side GEMMs
#define DEPTH_ 3

typedef __bf16 v16bf __attribute__((ext_vector_type(16)));
typedef __bf16 v8bf  __attribute__((ext_vector_type(8)));
typedef float  v8f   __attribute__((ext_vector_type(8)));

// ---------------------------------------------------------------------------
// Pack a row-major f32 weight [Krows x 256] (rows >= Krows are zero-padding)
// into bf16 B-fragment panels: Wp[((k/32)*256 + n)*32 + (k%32)] = W[k,n].
// With this layout a lane's 16 B-fragment values are two 16-byte runs (k 8h.., 16+8h..).
// ---------------------------------------------------------------------------
__global__ void pack_weights_kernel(const float* __restrict__ W,
                                    __hip_bfloat16* __restrict__ Wp,
                                    int Krows, int Kpad) {
  int i = blockIdx.x * blockDim.x + threadIdx.x;
  int total = Kpad * H_;
  if (i >= total) return;
  int k = i >> 8;          // / 256
  int n = i & 255;
  int ck = k >> 5;
  int r  = k & 31;
  float v = (k < Krows) ? W[(size_t)k * H_ + n] : 0.0f;
  Wp[((size_t)(ck * H_ + n)) * 32 + r] = __float2bfloat16(v);
}

// input_atom [25600, 82] f32 -> [25600, 96] bf16, zero padded K
__global__ void pack_input_atom_kernel(const float* __restrict__ ia,
                                       __hip_bfloat16* __restrict__ out) {
  int i = blockIdx.x * blockDim.x + threadIdx.x;
  if (i >= MN_ * 96) return;
  int m = i / 96, c = i % 96;
  float v = (c < AF_) ? ia[(size_t)m * AF_ + c] : 0.0f;
  out[i] = __float2bfloat16(v);
}

// ---------------------------------------------------------------------------
// bf16 WMMA GEMM: C[M x 256] = A[M x K] * Wp(K x 256, packed) (+bias)(relu?)
// Register-blocked: one wave owns a 64x64 output tile (4x4 accumulators of
// v8f). Per 32-wide K chunk: 8 A-frag loads + 8 B-frag loads -> 16 WMMAs.
// 8 waves per block. K must be a multiple of 32; M a multiple of 64.
// Epilogues:
//   outB != null : store bf16 (optionally bias+relu first)
//   outF != null : store f32  acc * fnMul[m,n] * rowMask[m]   (kernels epilogue)
// ---------------------------------------------------------------------------
__global__ __launch_bounds__(256)
void gemm_bf16_wmma_kernel(const __hip_bfloat16* __restrict__ A,
                           const __hip_bfloat16* __restrict__ Bp,
                           int Mrows, int Kdim,
                           const float* __restrict__ bias, int do_relu,
                           __hip_bfloat16* __restrict__ outB,
                           float* __restrict__ outF,
                           const float* __restrict__ fnMul,
                           const float* __restrict__ rowMask) {
  const int lane = threadIdx.x & 31;
  const int wave = threadIdx.x >> 5;
  const int NT   = H_ / 64;                  // 4 column supertiles
  int tile = blockIdx.x * 8 + wave;
  int mTiles = Mrows >> 6;                   // 64-row supertiles
  if (tile >= mTiles * NT) return;           // wave-uniform: EXEC stays full
  int mT = tile / NT, nT = tile % NT;
  int mBase = mT * 64, nBase = nT * 64;

  const int kHalf = lane >> 4;               // lane group selects K sub-block
  const int l15   = lane & 15;

  const __hip_bfloat16* pA[4];
  const __hip_bfloat16* pB[4];
#pragma unroll
  for (int i = 0; i < 4; ++i)
    pA[i] = A + (size_t)(mBase + i * 16 + l15) * Kdim;
#pragma unroll
  for (int j = 0; j < 4; ++j)
    pB[j] = Bp + (size_t)(nBase + j * 16 + l15) * 32 + (size_t)kHalf * 8;

  v8f acc[4][4] = {};
  const int nChunks = Kdim >> 5;

  union Frag { v16bf v; v8bf h[2]; };
  for (int c = 0; c < nChunks; ++c) {
    Frag aF[4], bF[4];
    // A fragment 16x32 bf16: lanes 0-15 rows M, K {0..7,16..23};
    //                        lanes 16-31 rows M, K {8..15,24..31}
#pragma unroll
    for (int i = 0; i < 4; ++i) {
      const __hip_bfloat16* p = pA[i] + c * 32;
      aF[i].h[0] = *(const v8bf*)(p + kHalf * 8);
      aF[i].h[1] = *(const v8bf*)(p + 16 + kHalf * 8);
    }
    // B fragment 32x16 bf16 from packed panels: K {0..7,16..23} / {8..15,24..31} per lane half
#pragma unroll
    for (int j = 0; j < 4; ++j) {
      const __hip_bfloat16* p = pB[j] + (size_t)c * (H_ * 32);
      bF[j].h[0] = *(const v8bf*)(p);
      bF[j].h[1] = *(const v8bf*)(p + 16);
    }
#pragma unroll
    for (int i = 0; i < 4; ++i)
#pragma unroll
      for (int j = 0; j < 4; ++j)
        acc[i][j] = __builtin_amdgcn_wmma_f32_16x16x32_bf16(
            false, aF[i].v, false, bF[j].v, (short)0, acc[i][j], false, false);
  }

#pragma unroll
  for (int j = 0; j < 4; ++j) {
    int nCol = nBase + j * 16 + l15;
    float bv = bias ? bias[nCol] : 0.0f;
#pragma unroll
    for (int i = 0; i < 4; ++i) {
#pragma unroll
      for (int r = 0; r < 8; ++r) {
        int m = mBase + i * 16 + r + kHalf * 8;  // C layout: VGPR r -> r / r+8
        float v = acc[i][j][r] + bv;
        if (do_relu) v = fmaxf(v, 0.0f);
        size_t idx = (size_t)m * H_ + nCol;
        if (outF) {
          outF[idx] = v * fnMul[idx] * rowMask[m];
        } else {
          outB[idx] = __float2bfloat16(v);
        }
      }
    }
  }
}

// ---------------------------------------------------------------------------
// Per-atom neighbor combine. One block per (molecule, atom); one thread per h.
// Bond-side projections (K-dim = 6) are done inline as scalar FMAs.
//   FN[m,h]          = sum_k mask * P[a_k,h] * (fbond_k . Wnb[:,h])
//   cat[m, 0..255]   = atom_features (bf16 passthrough, first half of concat)
//   cat[m, 256..511] = sum_k mask * relu(Q[a_k,h] + fbond_k . WU2bot[:,h])
// Q already contains b_U2. Neighbor indices are block-uniform -> scalar loads.
// ---------------------------------------------------------------------------
__global__ __launch_bounds__(256)
void gather_combine_kernel(const __hip_bfloat16* __restrict__ Pb,
                           const __hip_bfloat16* __restrict__ Qb,
                           const __hip_bfloat16* __restrict__ afb,
                           const float* __restrict__ input_bond,
                           const float* __restrict__ Wnb,      // [6,256]
                           const float* __restrict__ WU2,      // [262,256]
                           const int* __restrict__ atom_graph,
                           const int* __restrict__ bond_graph,
                           const int* __restrict__ num_nbs,
                           float* __restrict__ FN,
                           __hip_bfloat16* __restrict__ cat) {
  const int h  = threadIdx.x;                // 0..255
  const int bn = blockIdx.x;                 // 0..MN_-1
  const int b  = bn / N_;
  const int nnb = num_nbs[bn];

  float wnb_c[BF_], wu2_c[BF_];
#pragma unroll
  for (int c = 0; c < BF_; ++c) {
    wnb_c[c] = Wnb[c * H_ + h];
    wu2_c[c] = WU2[(H_ + c) * H_ + h];
  }

  float fn = 0.0f, nl = 0.0f;
  for (int k = 0; k < K_; ++k) {
    if (k >= nnb) break;                     // sequence mask (monotone)
    int ai = atom_graph[(size_t)bn * K_ + k];
    int bi = bond_graph[(size_t)bn * K_ + k];
    size_t arow = ((size_t)b * N_ + ai) * H_;
    const float* fb = input_bond + ((size_t)b * M_ + bi) * BF_;
    float sv = 0.0f, rv = 0.0f;
#pragma unroll
    for (int c = 0; c < BF_; ++c) {
      float bv = fb[c];
      sv += bv * wnb_c[c];
      rv += bv * wu2_c[c];
    }
    fn += __bfloat162float(Pb[arow + h]) * sv;
    float t = __bfloat162float(Qb[arow + h]) + rv;   // b_U2 folded into Q
    nl += fmaxf(t, 0.0f);
  }

  size_t m = (size_t)bn;
  FN[m * H_ + h]            = fn;
  cat[m * 2 * H_ + h]       = afb[m * H_ + h];
  cat[m * 2 * H_ + H_ + h]  = __float2bfloat16(nl);
}

// mol_hiddens[b,h] = sum_n kernels[b,n,h]
__global__ void mol_reduce_kernel(const float* __restrict__ kern,
                                  float* __restrict__ mol) {
  int i = blockIdx.x * blockDim.x + threadIdx.x;   // B*H
  if (i >= B_ * H_) return;
  int b = i >> 8, h = i & 255;
  const float* p = kern + ((size_t)b * N_) * H_ + h;
  float s = 0.0f;
  for (int n = 0; n < N_; ++n) s += p[(size_t)n * H_];
  mol[i] = s;
}

// ---------------------------------------------------------------------------
extern "C" void kernel_launch(void* const* d_in, const int* in_sizes, int n_in,
                              void* d_out, int out_size, void* d_ws, size_t ws_size,
                              hipStream_t stream) {
  (void)in_sizes; (void)n_in; (void)out_size;

  const float* input_atom = (const float*)d_in[0];
  const float* input_bond = (const float*)d_in[1];
  const float* node_mask  = (const float*)d_in[2];
  const float* W_embed    = (const float*)d_in[3];
  const float* W_nei_atom = (const float*)d_in[4];
  const float* W_nei_bond = (const float*)d_in[5];
  const float* W_self     = (const float*)d_in[6];
  const float* W_U2       = (const float*)d_in[7];
  const float* b_U2       = (const float*)d_in[8];
  const float* W_U1       = (const float*)d_in[9];
  const float* b_U1       = (const float*)d_in[10];
  const int*   atom_graph = (const int*)d_in[11];
  const int*   bond_graph = (const int*)d_in[12];
  const int*   num_nbs    = (const int*)d_in[13];

  float* out_kern = (float*)d_out;                    // [B,N,H]
  float* out_mol  = out_kern + (size_t)MN_ * H_;      // [B,H]

  // ---- workspace carving (256B aligned regions) ----
  char* ws = (char*)d_ws;
  size_t off = 0;
  auto carve = [&](size_t bytes) -> void* {
    void* p = ws + off;
    off = (off + bytes + 255) & ~(size_t)255;
    return p;
  };
  __hip_bfloat16* wemb_p = (__hip_bfloat16*)carve((size_t)96  * H_ * 2);
  __hip_bfloat16* wna_p  = (__hip_bfloat16*)carve((size_t)256 * H_ * 2);
  __hip_bfloat16* wself_p= (__hip_bfloat16*)carve((size_t)256 * H_ * 2);
  __hip_bfloat16* wu2t_p = (__hip_bfloat16*)carve((size_t)256 * H_ * 2);
  __hip_bfloat16* wu1_p  = (__hip_bfloat16*)carve((size_t)512 * H_ * 2);
  __hip_bfloat16* ia_b   = (__hip_bfloat16*)carve((size_t)MN_ * 96 * 2);
  __hip_bfloat16* af_b   = (__hip_bfloat16*)carve((size_t)MN_ * H_ * 2);
  __hip_bfloat16* p_b    = (__hip_bfloat16*)carve((size_t)MN_ * H_ * 2);
  __hip_bfloat16* q_b    = (__hip_bfloat16*)carve((size_t)MN_ * H_ * 2);
  __hip_bfloat16* cat_b  = (__hip_bfloat16*)carve((size_t)MN_ * 2 * H_ * 2);
  float*          fn_f   = (float*)carve((size_t)MN_ * H_ * 4);
  if (off > ws_size) return;   // insufficient scratch; bail safely

  const int tpb = 256;
  auto g = [&](int total) { return (total + tpb - 1) / tpb; };

  // ---- one-time packs (recomputed every call: deterministic) ----
  pack_weights_kernel<<<g(96  * H_), tpb, 0, stream>>>(W_embed,    wemb_p, AF_, 96);
  pack_weights_kernel<<<g(256 * H_), tpb, 0, stream>>>(W_nei_atom, wna_p,  256, 256);
  pack_weights_kernel<<<g(256 * H_), tpb, 0, stream>>>(W_self,     wself_p,256, 256);
  pack_weights_kernel<<<g(256 * H_), tpb, 0, stream>>>(W_U2,       wu2t_p, 256, 256); // top H rows
  pack_weights_kernel<<<g(512 * H_), tpb, 0, stream>>>(W_U1,       wu1_p,  512, 512);
  pack_input_atom_kernel<<<g(MN_ * 96), tpb, 0, stream>>>(input_atom, ia_b);

  // 64x64 tile per wave: (25600/64)*(256/64) = 1600 waves = 200 blocks
  const int gemmBlocks = (MN_ / 64) * (H_ / 64) / 8;

  // embed: af = relu(input_atom @ W_embed)
  gemm_bf16_wmma_kernel<<<gemmBlocks, 256, 0, stream>>>(
      ia_b, wemb_p, MN_, 96, nullptr, 1, af_b, nullptr, nullptr, nullptr);

  for (int d = 0; d < DEPTH_; ++d) {
    // P = af @ W_nei_atom
    gemm_bf16_wmma_kernel<<<gemmBlocks, 256, 0, stream>>>(
        af_b, wna_p, MN_, 256, nullptr, 0, p_b, nullptr, nullptr, nullptr);
    // Q = af @ W_U2_top + b_U2
    gemm_bf16_wmma_kernel<<<gemmBlocks, 256, 0, stream>>>(
        af_b, wu2t_p, MN_, 256, b_U2, 0, q_b, nullptr, nullptr, nullptr);
    // neighbor gather + bond projections + masked sums -> FN, cat
    gather_combine_kernel<<<MN_, 256, 0, stream>>>(
        p_b, q_b, af_b, input_bond, W_nei_bond, W_U2,
        atom_graph, bond_graph, num_nbs, fn_f, cat_b);
    // kernels = (af @ W_self) * FN * node_mask  -> d_out (last depth survives)
    gemm_bf16_wmma_kernel<<<gemmBlocks, 256, 0, stream>>>(
        af_b, wself_p, MN_, 256, nullptr, 0, nullptr, out_kern, fn_f, node_mask);
    // af = relu(cat @ W_U1 + b_U1)   (reads cat, safely overwrites af)
    gemm_bf16_wmma_kernel<<<gemmBlocks, 256, 0, stream>>>(
        cat_b, wu1_p, MN_, 512, b_U1, 1, af_b, nullptr, nullptr, nullptr);
  }

  mol_reduce_kernel<<<g(B_ * H_), tpb, 0, stream>>>(out_kern, out_mol);
}
